// SSN_17746804867732
// MI455X (gfx1250) — matrix operand order repaired and checked by reading, not compiled, not run on hardware
//
#include <hip/hip_runtime.h>
#include <cstddef>

// ---------------------------------------------------------------------------
// SSN / soft-SLIC on MI455X (gfx1250), wave32 + WMMA.
//   x:    (2, 20, 512, 512) f32
//   grid: 16x16 superpixels, each cell = 32x32 pixels, K=256, 9-neighborhood
//   out:  Q (2,9,512,512) | spf (2,20,256) | x passthrough
// Distances and Q-weighted aggregation both run on v_wmma_f32_16x16x32_f16;
// softmax shift-invariance removes |pf|^2; softmax lane reductions use
// batched v_permlane16_b32 steps (8-row ILP) and v_rcp_f32 normalization.
// ---------------------------------------------------------------------------

#define HH   512
#define CCH  20          // feature channels
#define CPAD 24          // padded: 20 feat + ch20 = ones (den) + 21..23 zero
#define NHW  16          // superpixel grid is 16x16
#define KSP  256
#define NEG_BIG -1.0e10f

typedef __attribute__((ext_vector_type(16))) _Float16 v16h;
typedef __attribute__((ext_vector_type(8)))  float    v8f;

union AFrag { v16h h; unsigned int u[8]; };

// XOR-permute within rows of 16 lanes via v_permlane16_b32 (pure VALU).
__device__ __forceinline__ float permx16(float v, int lo, int hi_) {
  const int s = __float_as_int(v);
  const int r = __builtin_amdgcn_permlane16(s, s, lo, hi_, false, false);
  return __int_as_float(r);
}

// --------------------------- init: per-cell means ---------------------------
__global__ __launch_bounds__(256) void ssn_init(const float* __restrict__ x,
                                                float* __restrict__ spf) {
  __shared__ float accs[CCH];
  const int cell = blockIdx.x;          // 0..255
  const int s    = blockIdx.y;          // sample
  const int cy = cell >> 4, cx = cell & 15;
  const int y0 = cy * 32, x0 = cx * 32;
  const int t = threadIdx.x;

  if (t < CCH) accs[t] = 0.f;
  __syncthreads();

  float a[CCH];
#pragma unroll
  for (int c = 0; c < CCH; ++c) a[c] = 0.f;
  for (int ii = 0; ii < 4; ++ii) {
    const int i  = t + ii * 256;        // pixel in 32x32 cell
    const int py = i >> 5, px = i & 31;
#pragma unroll
    for (int c = 0; c < CCH; ++c)
      a[c] += x[(((size_t)(s * CCH + c) * HH) + (y0 + py)) * HH + (x0 + px)];
  }
#pragma unroll
  for (int c = 0; c < CCH; ++c) {
    float v = a[c];
#pragma unroll
    for (int m = 16; m >= 1; m >>= 1) v += __shfl_xor(v, m, 32);  // wave32
    if ((t & 31) == 0) atomicAdd(&accs[c], v);
  }
  __syncthreads();
  if (t < CCH) spf[(s * KSP + cell) * CCH + t] = accs[t] * (1.0f / 1024.0f);
}

// --------------------------- zero accumulators ------------------------------
__global__ void ssn_zero(float* __restrict__ num, float* __restrict__ den) {
  const int t = blockIdx.x * 256 + threadIdx.x;
  if (t < 2 * KSP * CCH)                    num[t] = 0.f;
  else if (t < 2 * KSP * CCH + 2 * KSP)     den[t - 2 * KSP * CCH] = 0.f;
}

// ------------------- one E/M iteration: one workgroup / cell ----------------
__global__ __launch_bounds__(256) void ssn_iter(
    const float* __restrict__ x, const float* __restrict__ spf,
    float* __restrict__ num, float* __restrict__ den,
    float* __restrict__ qout, int write_q) {
  extern __shared__ char smem[];
  _Float16* PF    = (_Float16*)smem;             // [1024][24] f16 features
  float*    OUT   = (float*)(smem + 49152);      // [2][16][16] aggregation
  float*    SPFNP = (float*)(smem + 51200);      // [16][32] neighbor spf, 0-pad
  int*      NBR   = (int*)(smem + 53248);        // [16] neighbor cell index
  int*      VAL   = (int*)(smem + 53312);        // [16] validity (0 for j>=9)

  const int cell = blockIdx.x;
  const int s    = blockIdx.y;
  const int cy = cell >> 4, cx = cell & 15;
  const int y0 = cy * 32, x0 = cx * 32;
  const int t = threadIdx.x;

  if (t < 16) {
    int ok = 0, kk = 0;
    if (t < 9) {
      const int dy = t / 3 - 1, dx = t % 3 - 1;
      int ny = cy + dy, nx = cx + dx;
      ok = (ny >= 0 && ny < NHW && nx >= 0 && nx < NHW) ? 1 : 0;
      ny = min(max(ny, 0), NHW - 1);
      nx = min(max(nx, 0), NHW - 1);
      kk = ny * NHW + nx;
    }
    NBR[t] = kk; VAL[t] = ok;
  }
  OUT[t] = 0.f;   OUT[256 + t] = 0.f;
  SPFNP[t] = 0.f; SPFNP[256 + t] = 0.f;
  __syncthreads();

  if (t < 9 * CCH) {
    const int j = t / CCH, c = t % CCH;
    SPFNP[j * 32 + c] = spf[(s * KSP + NBR[j]) * CCH + c];
  }

  // Stage 32x32 pixel tile as f16 [pixel][24]; ch20 = 1.0 feeds the denominator.
  {
    const int i0 = t * 4;
    const int py = i0 >> 5, px = i0 & 31;
#pragma unroll
    for (int c = 0; c < CCH; ++c) {
      const float* gp = x + (((size_t)(s * CCH + c) * HH) + (y0 + py)) * HH + (x0 + px);
      if (c + 1 < CCH) __builtin_prefetch(gp + (size_t)HH * HH, 0, 1);
      const float4 v = *(const float4*)gp;
      PF[(i0 + 0) * CPAD + c] = (_Float16)v.x;
      PF[(i0 + 1) * CPAD + c] = (_Float16)v.y;
      PF[(i0 + 2) * CPAD + c] = (_Float16)v.z;
      PF[(i0 + 3) * CPAD + c] = (_Float16)v.w;
    }
#pragma unroll
    for (int q = 0; q < 4; ++q) {
      PF[(i0 + q) * CPAD + 20] = (_Float16)1.0f;
      PF[(i0 + q) * CPAD + 21] = (_Float16)0.0f;
      PF[(i0 + q) * CPAD + 22] = (_Float16)0.0f;
      PF[(i0 + q) * CPAD + 23] = (_Float16)0.0f;
    }
  }
  __syncthreads();

  const int wave = t >> 5, lane = t & 31;
  const int lj  = lane & 15;                 // B/C column (neighbor j); A row
  const int hi  = lane >> 4;                 // lane half selects K sub-ranges
  const int klo = hi ? 8 : 0;

  // Distance-GEMM B fragment: cols = neighbor j (zero-padded table -> no guards)
  v16h bdist;
  {
    const int kb = hi ? 16 : 0;
#pragma unroll
    for (int e2 = 0; e2 < 16; ++e2)
      bdist[e2] = (_Float16)SPFNP[lj * 32 + klo + e2 + ((e2 < 8) ? 0 : 8)];
  }
  // |spf_j|^2 per lane straight from the padded table (0 for j >= 9).
  float ssj = 0.f;
#pragma unroll
  for (int c = 0; c < CCH; ++c) {
    const float v = SPFNP[lj * 32 + c];
    ssj = fmaf(v, v, ssj);
  }
  const int okj = VAL[lj];

  v8f acc0 = {}; v8f acc1 = {};              // OUT[j][c0..15], OUT[j][c16..31]

#pragma unroll
  for (int pair = 0; pair < 4; ++pair) {
    const int tile0 = wave * 8 + pair * 2;   // 2 x 16-pixel tiles = one 32-K chunk
    AFrag aq;                                // Q^T fragment, built in-lane
#pragma unroll
    for (int half = 0; half < 2; ++half) {
      const int tile = tile0 + half;
      // A = pixel features: rows = 16 pixels, K = 32 padded channels.
      // Out-of-row K pairs are clamped onto the zero pad (halves 22/23):
      AFrag af;
      const unsigned int* rowp = (const unsigned int*)(PF + (tile * 16 + lj) * CPAD);
#pragma unroll
      for (int v = 0; v < 8; ++v) {
        const int kp  = klo + 2 * v + ((v < 4) ? 0 : 8);
        const int kpc = (kp < CPAD) ? kp : 22;     // clamp -> zero channels
        af.u[v] = rowp[kpc >> 1];
      }
      v8f cf = {};
      cf = __builtin_amdgcn_wmma_f32_16x16x32_f16(false, af.h, false, bdist,
                                                  (short)0, cf, false, false);

      // Batched softmax over j: 8 independent rows per reduction step for ILP.
      float valv[8], mx[8], ex[8], sm[8];
#pragma unroll
      for (int r = 0; r < 8; ++r) {
        valv[r] = okj ? fmaf(2.0f, cf[r], -ssj) : NEG_BIG;
        mx[r] = valv[r];
      }
#pragma unroll
      for (int r = 0; r < 8; ++r) mx[r] = fmaxf(mx[r], permx16(mx[r], 0x67452301, 0xEFCDAB89));
#pragma unroll
      for (int r = 0; r < 8; ++r) mx[r] = fmaxf(mx[r], permx16(mx[r], 0x54761032, 0xDCFE98BA));
#pragma unroll
      for (int r = 0; r < 8; ++r) mx[r] = fmaxf(mx[r], permx16(mx[r], 0x32107654, 0xBA98FEDC));
#pragma unroll
      for (int r = 0; r < 8; ++r) mx[r] = fmaxf(mx[r], permx16(mx[r], 0xFEDCBA98, 0x76543210));
#pragma unroll
      for (int r = 0; r < 8; ++r) { ex[r] = __expf(valv[r] - mx[r]); sm[r] = ex[r]; }
#pragma unroll
      for (int r = 0; r < 8; ++r) sm[r] += permx16(sm[r], 0x67452301, 0xEFCDAB89);
#pragma unroll
      for (int r = 0; r < 8; ++r) sm[r] += permx16(sm[r], 0x54761032, 0xDCFE98BA);
#pragma unroll
      for (int r = 0; r < 8; ++r) sm[r] += permx16(sm[r], 0x32107654, 0xBA98FEDC);
#pragma unroll
      for (int r = 0; r < 8; ++r) sm[r] += permx16(sm[r], 0xFEDCBA98, 0x76543210);
#pragma unroll
      for (int r = 0; r < 8; ++r) {
        // sm in [1,16]: v_rcp_f32 + mul is safe and co-executes (TRANS).
        const float q = ex[r] * __builtin_amdgcn_rcpf(sm[r]);
        aq.h[half * 8 + r] = (_Float16)q;    // lands exactly where GEMM2 A wants it
        if (write_q && lj < 9) {
          const int pp = tile * 16 + r + hi * 8;
          const int yy = y0 + (pp >> 5), xx = x0 + (pp & 31);
          qout[((size_t)(s * 9 + lj) * HH + yy) * HH + xx] = q;
        }
      }
    }
    // Aggregation GEMM: OUT[j][c] += Q^T x PF over this 32-pixel K chunk.
    const int kb0 = tile0 * 16 + hi * 8;
    const int c1  = (16 + lj < CPAD) ? (16 + lj) : 21;   // clamp -> zero channel
    v16h b0, b1;
#pragma unroll
    for (int e2 = 0; e2 < 16; ++e2) {
      b0[e2] = PF[(kb0 + e2 + ((e2 < 8) ? 0 : 8)) * CPAD + lj];
      b1[e2] = PF[(kb0 + e2 + ((e2 < 8) ? 0 : 8)) * CPAD + c1];
    }
    acc0 = __builtin_amdgcn_wmma_f32_16x16x32_f16(false, aq.h, false, b0,
                                                  (short)0, acc0, false, false);
    acc1 = __builtin_amdgcn_wmma_f32_16x16x32_f16(false, aq.h, false, b1,
                                                  (short)0, acc1, false, false);
  }

  // Reduce per-wave partials across the 8 waves in LDS.
#pragma unroll
  for (int r = 0; r < 8; ++r) {
    const int row = r + hi * 8;
    atomicAdd(&OUT[(0 * 16 + row) * 16 + lj], acc0[r]);
    atomicAdd(&OUT[(1 * 16 + row) * 16 + lj], acc1[r]);
  }
  __syncthreads();

  // Push 9 neighbors x (20 channels + den) to global accumulators.
  if (t < 9 * (CCH + 1)) {
    const int j  = t / (CCH + 1);
    const int cc = t % (CCH + 1);
    const float v = OUT[((cc >> 4) * 16 + j) * 16 + (cc & 15)];
    const int k = NBR[j];
    if (cc < CCH) atomicAdd(&num[(s * KSP + k) * CCH + cc], v);
    else          atomicAdd(&den[s * KSP + k], v);      // ones-channel
  }
}

// --------------------------- spf = num / max(den) ---------------------------
__global__ void ssn_update(const float* __restrict__ num,
                           const float* __restrict__ den,
                           float* __restrict__ spf,
                           float* __restrict__ spf_out, int last) {
  const int t = blockIdx.x * 256 + threadIdx.x;
  if (t >= 2 * KSP * CCH) return;
  const int s = t / (KSP * CCH);
  const int r = t % (KSP * CCH);
  const int k = r / CCH, c = r % CCH;
  const float d = den[s * KSP + k];
  const float v = num[t] / fmaxf(d, 1e-8f);
  spf[t] = v;
  if (last) spf_out[(s * CCH + c) * KSP + k] = v;       // (b, c, K) layout
}

// ---------------------------------------------------------------------------
extern "C" void kernel_launch(void* const* d_in, const int* in_sizes, int n_in,
                              void* d_out, int out_size, void* d_ws, size_t ws_size,
                              hipStream_t stream) {
  (void)in_sizes; (void)n_in; (void)out_size; (void)ws_size;
  const float* x = (const float*)d_in[0];

  float* out     = (float*)d_out;
  float* qout    = out;                                   // 2*9*512*512
  float* spf_out = out + (size_t)2 * 9 * HH * HH;         // 2*20*256
  float* x_out   = spf_out + (size_t)2 * CCH * KSP;       // 2*20*512*512

  float* spf = (float*)d_ws;                              // 10240 f32
  float* num = spf + 2 * KSP * CCH;                       // 10240 f32
  float* den = num + 2 * KSP * CCH;                       // 512 f32

  ssn_init<<<dim3(KSP, 2), 256, 0, stream>>>(x, spf);

  const size_t smem = 53376;                              // < 64 KB
  for (int it = 0; it < 10; ++it) {
    const int last = (it == 9);
    ssn_zero<<<42, 256, 0, stream>>>(num, den);
    ssn_iter<<<dim3(KSP, 2), 256, smem, stream>>>(x, spf, num, den, qout, last);
    ssn_update<<<40, 256, 0, stream>>>(num, den, spf, spf_out, last);
  }

  hipMemcpyAsync(x_out, x, (size_t)2 * CCH * HH * HH * sizeof(float),
                 hipMemcpyDeviceToDevice, stream);
}
